// GlobalLoss_17566416240906
// MI455X (gfx1250) — matrix operand order repaired and checked by reading, not compiled, not run on hardware
//
#include <hip/hip_runtime.h>
#include <stdint.h>

typedef __attribute__((ext_vector_type(16))) _Float16 v16h;
typedef __attribute__((ext_vector_type(8)))  float    v8f;

#define NBINS 65536
static const float EPSF = 2.2204460492503131e-16f;

__device__ __forceinline__ unsigned f2k(float f) {
    unsigned u = __float_as_uint(f);
    return (u & 0x80000000u) ? ~u : (u | 0x80000000u);
}
__device__ __forceinline__ float k2f(unsigned k) {
    unsigned u = (k & 0x80000000u) ? (k & 0x7FFFFFFFu) : ~k;
    return __uint_as_float(u);
}

// ---------------- init: zero histogram, set flag=1 ----------------
__global__ void k_init(unsigned* __restrict__ hist, int nbins, int* __restrict__ flag) {
    if (blockIdx.x == 0 && threadIdx.x == 0) *flag = 1;
    for (int i = blockIdx.x * blockDim.x + threadIdx.x; i < nbins;
         i += gridDim.x * blockDim.x)
        hist[i] = 0u;
}

// Pack 16 consecutive f32 into WMMA A-operand; k order used here (same for B) per
// lane: halfwords 0..15 = K kbase..kbase+15 (kbase = 16*(lane/16)).
__device__ __forceinline__ v16h pack_a(const float* __restrict__ ap0) {
    const float4 q0 = ((const float4*)ap0)[0];
    const float4 q1 = ((const float4*)ap0)[1];
    const float4 q2 = ((const float4*)ap0)[2];
    const float4 q3 = ((const float4*)ap0)[3];
    v16h A;
    A[0]  = (_Float16)q0.x; A[1]  = (_Float16)q0.y; A[2]  = (_Float16)q0.z; A[3]  = (_Float16)q0.w;
    A[4]  = (_Float16)q1.x; A[5]  = (_Float16)q1.y; A[6]  = (_Float16)q1.z; A[7]  = (_Float16)q1.w;
    A[8]  = (_Float16)q2.x; A[9]  = (_Float16)q2.y; A[10] = (_Float16)q2.z; A[11] = (_Float16)q2.w;
    A[12] = (_Float16)q3.x; A[13] = (_Float16)q3.y; A[14] = (_Float16)q3.z; A[15] = (_Float16)q3.w;
    return A;
}
__device__ __forceinline__ v16h pack_b(const float* __restrict__ bp) {
    const float4 r0 = ((const float4*)bp)[0];
    const float4 r1 = ((const float4*)bp)[1];
    const float4 r2 = ((const float4*)bp)[2];
    const float4 r3 = ((const float4*)bp)[3];
    v16h B;
    B[0]  = (_Float16)r0.x; B[1]  = (_Float16)r0.y; B[2]  = (_Float16)r0.z; B[3]  = (_Float16)r0.w;
    B[4]  = (_Float16)r1.x; B[5]  = (_Float16)r1.y; B[6]  = (_Float16)r1.z; B[7]  = (_Float16)r1.w;
    B[8]  = (_Float16)r2.x; B[9]  = (_Float16)r2.y; B[10] = (_Float16)r2.z; B[11] = (_Float16)r2.w;
    B[12] = (_Float16)r3.x; B[13] = (_Float16)r3.y; B[14] = (_Float16)r3.z; B[15] = (_Float16)r3.w;
    return B;
}

// ---------------- phase 1: GEMV via v_wmma_f32_16x16x32_f16 ----------------
// block = 256 threads = 8 waves; wave handles TWO 16-row strips (32 rows),
// sharing the B operand (w chunk replicated across all 16 columns) between
// two independent WMMA chains. Block covers 256 rows.
// B k order used here (32x16, same as A): lanes 0-15 hold K=0..15, lanes 16-31 hold K=16..31 ->
// each lane loads 16 consecutive w values.
template <int DFIX>
__global__ void k_gemv_wmma(const float* __restrict__ x, const float* __restrict__ w,
                            const float* __restrict__ bias, float* __restrict__ out,
                            float* __restrict__ sum_partials, int* __restrict__ flag,
                            int N, int Drt) {
    const int D     = DFIX ? DFIX : Drt;
    const int lane  = threadIdx.x & 31;
    const int wid   = threadIdx.x >> 5;
    const int rows  = blockIdx.x * 256 + wid * 32;
    const int m     = lane & 15;
    const int kbase = (lane < 16) ? 0 : 16;
    const float* xrow0 = x + (size_t)(rows + m) * (size_t)D;
    const float* xrow1 = xrow0 + (size_t)16 * (size_t)D;
    const float* wb    = w + ((lane < 16) ? 0 : 16);

    v8f c0 = {}, c1 = {};
#pragma unroll 4
    for (int k0 = 0; k0 < D; k0 += 32) {
        const float* ap0 = xrow0 + k0 + kbase;
        const float* ap1 = xrow1 + k0 + kbase;
        // WGP-scope prefetch of the next K-chunk of both row strips
        // (locality 3 -> near scope: fill all cache levels)
        __builtin_prefetch(ap0 + 32, 0, 3);
        __builtin_prefetch(ap1 + 32, 0, 3);
        const v16h A0 = pack_a(ap0);
        const v16h A1 = pack_a(ap1);
        const v16h B  = pack_b(wb + k0);
        c0 = __builtin_amdgcn_wmma_f32_16x16x32_f16(false, A0, false, B,
                                                    (short)0, c0, false, false);
        c1 = __builtin_amdgcn_wmma_f32_16x16x32_f16(false, A1, false, B,
                                                    (short)0, c1, false, false);
    }

    // D layout: VGPR r, lanes 0-15 -> M=r, N=lane; lanes 16-31 -> M=8+r, N=lane-16.
    // All columns equal; lanes 0 and 16 carry column 0 for rows 0-7 / 8-15.
    __shared__ float s_sum[16];
    __shared__ int   s_ok[16];
    if (m == 0) {
        const float bv = bias[0];
        const int rbase0 = rows + ((lane == 16) ? 8 : 0);
        const int rbase1 = rbase0 + 16;
        float ls = 0.0f;
        int ok = 1;
#pragma unroll
        for (int r = 0; r < 8; ++r) {
            float v0 = c0[r] + bv;
            float v1 = c1[r] + bv;
            out[rbase0 + r] = v0;
            out[rbase1 + r] = v1;
            ls += v0 + v1;
            ok &= (v0 + EPSF > 0.0f) ? 1 : 0;
            ok &= (v1 + EPSF > 0.0f) ? 1 : 0;
        }
        const int slot = wid * 2 + ((lane == 16) ? 1 : 0);
        s_sum[slot] = ls;
        s_ok[slot]  = ok;
    }
    __syncthreads();
    if (threadIdx.x == 0) {
        float t = 0.0f;
        int ok = 1;
        for (int i = 0; i < 16; ++i) { t += s_sum[i]; ok &= s_ok[i]; }
        sum_partials[blockIdx.x] = t;
        if (!ok) atomicAnd(flag, 0);
    }
}

// ---------------- phase 2a: 65536-bin histogram of high key bits ----------------
__global__ void k_hist(const float* __restrict__ out, unsigned* __restrict__ hist, int N) {
    for (int i = blockIdx.x * blockDim.x + threadIdx.x; i < N;
         i += gridDim.x * blockDim.x)
        atomicAdd(&hist[f2k(out[i]) >> 16], 1u);
}

// ---------------- phase 2b: exclusive scan of 65536 bins (1 block, 1024 thr) ----
__global__ void k_scan(const unsigned* __restrict__ hist, unsigned* __restrict__ base,
                       unsigned* __restrict__ cursor) {
    __shared__ unsigned tsum[1024];
    __shared__ unsigned tpre[1024];
    const int t = threadIdx.x;
    unsigned s = 0;
    for (int i = 0; i < 64; ++i) s += hist[t * 64 + i];
    tsum[t] = s;
    __syncthreads();
    if (t == 0) {
        unsigned run = 0;
        for (int i = 0; i < 1024; ++i) { tpre[i] = run; run += tsum[i]; }
    }
    __syncthreads();
    unsigned run = tpre[t];
    for (int i = 0; i < 64; ++i) {
        const int idx = t * 64 + i;
        base[idx]   = run;
        cursor[idx] = run;
        run += hist[idx];
    }
}

// ---------------- phase 2c: scatter keys into buckets (order in bucket free) ----
__global__ void k_scatter(const float* __restrict__ out, unsigned* __restrict__ cursor,
                          unsigned* __restrict__ sorted, int N) {
    for (int i = blockIdx.x * blockDim.x + threadIdx.x; i < N;
         i += gridDim.x * blockDim.x) {
        const unsigned k = f2k(out[i]);
        const unsigned pos = atomicAdd(&cursor[k >> 16], 1u);
        sorted[pos] = k;
    }
}

// ---------------- phase 2d: exact rank within bucket -> pair_sum partials ----
// pair_sum = sum_i out_i * (2*lt_i + eq_i - n)  (tie-symmetric exact form)
__global__ void k_rank(const unsigned* __restrict__ sorted, const unsigned* __restrict__ base,
                       const unsigned* __restrict__ hist, float* __restrict__ pair_partials,
                       int N) {
    float acc = 0.0f;
    for (int p = blockIdx.x * blockDim.x + threadIdx.x; p < N;
         p += gridDim.x * blockDim.x) {
        const unsigned k  = sorted[p];
        const unsigned hi = k >> 16;
        const unsigned s  = base[hi];
        const unsigned cnt = hist[hi];
        unsigned lt = 0, eq = 0;
        for (unsigned q = 0; q < cnt; ++q) {
            const unsigned kq = sorted[s + q];
            lt += (kq < k) ? 1u : 0u;
            eq += (kq == k) ? 1u : 0u;
        }
        const float coef = 2.0f * (float)(s + lt) + (float)eq - (float)N;
        acc += k2f(k) * coef;
    }
    __shared__ float sh[256];
    sh[threadIdx.x] = acc;
    __syncthreads();
    for (int off = 128; off > 0; off >>= 1) {
        if (threadIdx.x < off) sh[threadIdx.x] += sh[threadIdx.x + off];
        __syncthreads();
    }
    if (threadIdx.x == 0) pair_partials[blockIdx.x] = sh[0];
}

// ---------------- finalize: deterministic fixed-order reductions ----------------
__global__ void k_final(const float* __restrict__ sum_partials, int nsum,
                        const float* __restrict__ pair_partials, int npair,
                        const int* __restrict__ flag, float* __restrict__ out2,
                        float reg_over_n) {
    __shared__ float sh[256];
    __shared__ float total_sum;
    float a = 0.0f;
    for (int i = threadIdx.x; i < nsum; i += 256) a += sum_partials[i];
    sh[threadIdx.x] = a;
    __syncthreads();
    for (int off = 128; off > 0; off >>= 1) {
        if (threadIdx.x < off) sh[threadIdx.x] += sh[threadIdx.x + off];
        __syncthreads();
    }
    if (threadIdx.x == 0) total_sum = sh[0];
    __syncthreads();
    float p = 0.0f;
    for (int i = threadIdx.x; i < npair; i += 256) p += pair_partials[i];
    sh[threadIdx.x] = p;
    __syncthreads();
    for (int off = 128; off > 0; off >>= 1) {
        if (threadIdx.x < off) sh[threadIdx.x] += sh[threadIdx.x + off];
        __syncthreads();
    }
    if (threadIdx.x == 0) {
        out2[0] = -total_sum + reg_over_n * sh[0];
        out2[1] = (*flag) ? 1.0f : 0.0f;
    }
}

extern "C" void kernel_launch(void* const* d_in, const int* in_sizes, int n_in,
                              void* d_out, int out_size, void* d_ws, size_t ws_size,
                              hipStream_t stream) {
    const float* x = (const float*)d_in[0];
    const float* w = (const float*)d_in[1];
    const float* b = (const float*)d_in[2];
    float* outv = (float*)d_out;

    const int D = in_sizes[1];
    const int N = in_sizes[0] / D;   // 262144 x 512

    // workspace carve (256B aligned)
    char* ws = (char*)d_ws;
    size_t off = 0;
    auto carve = [&](size_t bytes) -> void* {
        off = (off + 255) & ~(size_t)255;
        void* p = ws + off;
        off += bytes;
        return p;
    };
    float*    out_f         = (float*)carve((size_t)N * 4);
    unsigned* sorted        = (unsigned*)carve((size_t)N * 4);
    unsigned* hist          = (unsigned*)carve((size_t)NBINS * 4);
    unsigned* base          = (unsigned*)carve((size_t)NBINS * 4);
    unsigned* cursor        = (unsigned*)carve((size_t)NBINS * 4);
    const int gemvBlocks    = N / 256;   // 32 rows/wave * 8 waves
    float*    sum_partials  = (float*)carve((size_t)gemvBlocks * 4);
    const int rankBlocks    = 1024;
    float*    pair_partials = (float*)carve((size_t)rankBlocks * 4);
    int*      flag          = (int*)carve(4);

    k_init<<<64, 256, 0, stream>>>(hist, NBINS, flag);
    if (D == 512) {
        k_gemv_wmma<512><<<gemvBlocks, 256, 0, stream>>>(x, w, b, out_f, sum_partials,
                                                         flag, N, D);
    } else {
        k_gemv_wmma<0><<<gemvBlocks, 256, 0, stream>>>(x, w, b, out_f, sum_partials,
                                                       flag, N, D);
    }
    k_hist<<<512, 256, 0, stream>>>(out_f, hist, N);
    k_scan<<<1, 1024, 0, stream>>>(hist, base, cursor);
    k_scatter<<<512, 256, 0, stream>>>(out_f, cursor, sorted, N);
    k_rank<<<1024, 256, 0, stream>>>(sorted, base, hist, pair_partials, N);
    k_final<<<1, 256, 0, stream>>>(sum_partials, gemvBlocks, pair_partials, 1024,
                                   flag, outv, 0.1f / (float)N);
}
